// SchNet_45457933860992
// MI455X (gfx1250) — matrix operand order repaired and checked by reading, not compiled, not run on hardware
//
#include <hip/hip_runtime.h>

// ---------------------------------------------------------------------------
// SchNet forward on gfx1250 (MI455X): all GEMMs via v_wmma_f32_16x16x32_bf16
// ---------------------------------------------------------------------------

typedef __attribute__((ext_vector_type(16))) __bf16 v16bf;
typedef __attribute__((ext_vector_type(8)))  __bf16 v8bf;
typedef __attribute__((ext_vector_type(8)))  float  v8f;

#define LN2F    0.6931471805599453f
#define CUTOFF  10.0f
#define BOXF    5.0f
#define PI_F    3.14159265358979323846f

// ShiftedSoftplus, numerically stable
__device__ __forceinline__ float ssp(float x) {
  return fmaxf(x, 0.f) + log1pf(__expf(-fabsf(x))) - LN2F;
}

__device__ __forceinline__ v8f splat8(float x) {
  v8f t;
#pragma unroll
  for (int r = 0; r < 8; ++r) t[r] = x;
  return t;
}

// A fragment, 16x32 bf16 (MxK). Lane L: row M = L&15,
// K = {kb..kb+7} U {kb+16..kb+23}, kb = k0 + 8*(L>=16).  (ISA 7.12.2)
__device__ __forceinline__ v16bf load_a_frag(const __bf16* tile, int ld,
                                             int m, int k0, int lane) {
  const int kb = k0 + ((lane >> 4) << 3);
  const __bf16* p = tile + (size_t)m * ld + kb;
  v8bf lo = *(const v8bf*)p;
  v8bf hi = *(const v8bf*)(p + 16);
  v16bf a;
#pragma unroll
  for (int t = 0; t < 8; ++t) { a[t] = lo[t]; a[t + 8] = hi[t]; }
  return a;
}

// B fragment, 32x16 bf16 (KxN), weights stored transposed wt[n][k].
// Lane L: col N = n0 + (L&15), K = {kb..kb+7} U {kb+16..kb+23}, kb = k0 + 8*(L>=16).
__device__ __forceinline__ v16bf load_b_frag(const __bf16* wt, int ld,
                                             int n0, int k0, int lane) {
  const int n = n0 + (lane & 15);
  const int k = k0 + ((lane >> 4) << 3);
  const __bf16* p = wt + (size_t)n * ld + k;
  v8bf lo = *(const v8bf*)p;
  v8bf hi = *(const v8bf*)(p + 16);
  v16bf b;
#pragma unroll
  for (int t = 0; t < 8; ++t) { b[t] = lo[t]; b[t + 8] = hi[t]; }
  return b;
}

__device__ __forceinline__ v8f wmma_bf16(v16bf a, v16bf b, v8f c) {
  return __builtin_amdgcn_wmma_f32_16x16x32_bf16(false, a, false, b,
                                                 (short)0, c, false, false);
}

// One wave computes a 16 x (16*NT) strip over K = KD (chunks of 32).
// All NT B-fragments for a k-chunk are loaded up front (batched into one
// load clause) so the NT WMMAs issue back-to-back behind a single wait.
template <int KD, int NT>
__device__ __forceinline__ void gemm_strip(const __bf16* A, int lda,
                                           const __bf16* Bt, int ldb,
                                           int m0, int lane, v8f acc[NT]) {
#pragma unroll
  for (int k0 = 0; k0 < KD; k0 += 32) {
    v16bf a = load_a_frag(A, lda, m0 + (lane & 15), k0, lane);
    v16bf b[NT];
#pragma unroll
    for (int j = 0; j < NT; ++j) b[j] = load_b_frag(Bt, ldb, j * 16, k0, lane);
#pragma unroll
    for (int j = 0; j < NT; ++j) acc[j] = wmma_bf16(a, b[j], acc[j]);
  }
}

// ---------------------------------------------------------------------------
// Prep: convert all weights to bf16, transposed to [out][in]; pad G 50->64.
// ---------------------------------------------------------------------------
__global__ void __launch_bounds__(256)
prep_weights(const float* mw1, const float* mw2, const float* cw1,
             const float* cw2, const float* lw, const float* ow1,
             __bf16* w1t, __bf16* w2t, __bf16* cf1t, __bf16* cf2t,
             __bf16* lwt, __bf16* ow1t) {
  const int S0 = 6 * 128 * 64;    // w1t  [L][F=128][K=64]
  const int S1 = 6 * 128 * 128;   // each square [L][out=128][in=128]
  int i = blockIdx.x * blockDim.x + threadIdx.x;
  if (i < S0) {
    int l = i / (128 * 64), rem = i % (128 * 64);
    int f = rem / 64, g = rem % 64;
    w1t[i] = (__bf16)((g < 50) ? mw1[((size_t)l * 50 + g) * 128 + f] : 0.f);
    return;
  }
  i -= S0;
  if (i < S1) { int l = i >> 14, o = (i >> 7) & 127, in = i & 127;
    w2t[i]  = (__bf16)mw2[((size_t)l << 14) + (in << 7) + o]; return; }
  i -= S1;
  if (i < S1) { int l = i >> 14, o = (i >> 7) & 127, in = i & 127;
    cf1t[i] = (__bf16)cw1[((size_t)l << 14) + (in << 7) + o]; return; }
  i -= S1;
  if (i < S1) { int l = i >> 14, o = (i >> 7) & 127, in = i & 127;
    cf2t[i] = (__bf16)cw2[((size_t)l << 14) + (in << 7) + o]; return; }
  i -= S1;
  if (i < S1) { int l = i >> 14, o = (i >> 7) & 127, in = i & 127;
    lwt[i]  = (__bf16)lw[((size_t)l << 14) + (in << 7) + o]; return; }
  i -= S1;
  if (i < 64 * 128) { int o = i >> 7, in = i & 127;
    ow1t[i] = (__bf16)ow1[in * 64 + o]; return; }
}

// ---------------------------------------------------------------------------
// Edge geometry: PBC distance, Gaussian smearing (bf16, padded), cosine cutoff
// ---------------------------------------------------------------------------
__global__ void __launch_bounds__(256)
edge_geom(const float* pos, const int* rowi, const int* coli,
          __bf16* ea, float* Cbuf, int E_) {
  int e = blockIdx.x * blockDim.x + threadIdx.x;
  if (e >= E_) return;
  int r = rowi[e], c = coli[e];
  float dx = pos[r * 3 + 0] - pos[c * 3 + 0];
  float dy = pos[r * 3 + 1] - pos[c * 3 + 1];
  float dz = pos[r * 3 + 2] - pos[c * 3 + 2];
  dx -= rintf(dx * (1.f / BOXF)) * BOXF;
  dy -= rintf(dy * (1.f / BOXF)) * BOXF;
  dz -= rintf(dz * (1.f / BOXF)) * BOXF;
  float ew = sqrtf(dx * dx + dy * dy + dz * dz);
  const float delta = CUTOFF / 49.f;
  const float coeff = -0.5f / (delta * delta);
  __bf16* row = ea + (size_t)e * 64;
#pragma unroll 1
  for (int g = 0; g < 50; ++g) {
    float d = ew - (float)g * delta;
    row[g] = (__bf16)__expf(coeff * d * d);
  }
#pragma unroll
  for (int g = 50; g < 64; ++g) row[g] = (__bf16)0.f;
  float C = 0.5f * (__cosf(ew * (PI_F / CUTOFF)) + 1.f);
  Cbuf[e] = (ew < CUTOFF) ? C : 0.f;
}

// ---------------------------------------------------------------------------
__global__ void __launch_bounds__(256)
emb_init(const int* z, const float* emb, float* h, __bf16* hb, int total) {
  int i = blockIdx.x * blockDim.x + threadIdx.x;
  if (i >= total) return;
  int n = i >> 7, c = i & 127;
  float v = emb[z[n] * 128 + c];
  h[i] = v; hb[i] = (__bf16)v;
}

__global__ void __launch_bounds__(256) zero_f(float* p, int n) {
  int i = blockIdx.x * blockDim.x + threadIdx.x;
  if (i < n) p[i] = 0.f;
}

// ---------------------------------------------------------------------------
// xh = h @ cf_w1[l]   (no bias).  128-node tile, 8 waves x 16 rows.
// ---------------------------------------------------------------------------
__global__ void __launch_bounds__(256)
node_lin1(const __bf16* hb, const __bf16* cf1t, float* xh) {
  const int tid = threadIdx.x, lane = tid & 31, wave = tid >> 5;
  const int m0 = wave * 16, ln = lane & 15, hi8 = (lane >> 4) << 3;
  const int t0 = blockIdx.x * 128;
  v8f acc[8];
#pragma unroll
  for (int j = 0; j < 8; ++j) acc[j] = splat8(0.f);
  gemm_strip<128, 8>(hb + (size_t)t0 * 128, 128, cf1t, 128, m0, lane, acc);
#pragma unroll
  for (int j = 0; j < 8; ++j)
#pragma unroll
    for (int r = 0; r < 8; ++r)
      xh[(size_t)(t0 + m0 + hi8 + r) * 128 + j * 16 + ln] = acc[j][r];
}

// ---------------------------------------------------------------------------
// Fused edge filter + message scatter:
//   W = ssp(ea@w1+b1)@w2+b2, W *= C;  agg[col] += xh[row] * W  (fp32 atomics)
// ---------------------------------------------------------------------------
__global__ void __launch_bounds__(256)
edge_filter_scatter(const __bf16* ea, const float* Cbuf,
                    const __bf16* w1t, const float* b1,
                    const __bf16* w2t, const float* b2,
                    const float* xh, const int* rowi, const int* coli,
                    float* agg) {
  __shared__ __bf16 tile[128 * 128];
  const int tid = threadIdx.x, lane = tid & 31, wave = tid >> 5;
  const int m0 = wave * 16, ln = lane & 15, hi8 = (lane >> 4) << 3;
  const int e0 = blockIdx.x * 128;

  // GEMM1: [128x64] @ [64x128] + b1, ssp -> bf16 LDS
  v8f acc[8];
#pragma unroll
  for (int j = 0; j < 8; ++j) acc[j] = splat8(b1[j * 16 + ln]);
  gemm_strip<64, 8>(ea + (size_t)e0 * 64, 64, w1t, 64, m0, lane, acc);
#pragma unroll
  for (int j = 0; j < 8; ++j)
#pragma unroll
    for (int r = 0; r < 8; ++r)
      tile[(m0 + hi8 + r) * 128 + j * 16 + ln] = (__bf16)ssp(acc[j][r]);
  __syncthreads();

  // GEMM2: [128x128] @ [128x128] + b2
  v8f acc2[8];
#pragma unroll
  for (int j = 0; j < 8; ++j) acc2[j] = splat8(b2[j * 16 + ln]);
  gemm_strip<128, 8>(tile, 128, w2t, 128, m0, lane, acc2);

  // cutoff scale + gather xh[row] + scatter-add to agg[col]
  int er[8], ec[8]; float cc[8];
#pragma unroll
  for (int r = 0; r < 8; ++r) {
    int e = e0 + m0 + hi8 + r;
    er[r] = rowi[e]; ec[r] = coli[e]; cc[r] = Cbuf[e];
  }
#pragma unroll
  for (int j = 0; j < 8; ++j) {
    const int n = j * 16 + ln;
#pragma unroll
    for (int r = 0; r < 8; ++r) {
      float w = acc2[j][r] * cc[r];
      float v = w * xh[(size_t)er[r] * 128 + n];
      atomicAdd(&agg[(size_t)ec[r] * 128 + n], v);
    }
  }
}

// ---------------------------------------------------------------------------
// Node update: h += ssp(agg@cf_w2 + cf_b2) @ lin_w + lin_b   (single LDS tile
// reused across the two GEMMs; also refresh the bf16 shadow of h)
// ---------------------------------------------------------------------------
__global__ void __launch_bounds__(256)
node_update(const float* agg, const __bf16* cf2t, const float* cb2,
            const __bf16* lwt, const float* lb, float* h, __bf16* hb) {
  __shared__ __bf16 tile[128 * 128];
  const int tid = threadIdx.x, lane = tid & 31, wave = tid >> 5;
  const int m0 = wave * 16, ln = lane & 15, hi8 = (lane >> 4) << 3;
  const int t0 = blockIdx.x * 128;

  for (int i = tid; i < 128 * 128; i += 256)
    tile[i] = (__bf16)agg[(size_t)t0 * 128 + i];
  __syncthreads();

  v8f acc[8];
#pragma unroll
  for (int j = 0; j < 8; ++j) acc[j] = splat8(cb2[j * 16 + ln]);
  gemm_strip<128, 8>(tile, 128, cf2t, 128, m0, lane, acc);
  __syncthreads();   // all reads of tile complete before reuse
#pragma unroll
  for (int j = 0; j < 8; ++j)
#pragma unroll
    for (int r = 0; r < 8; ++r)
      tile[(m0 + hi8 + r) * 128 + j * 16 + ln] = (__bf16)ssp(acc[j][r]);
  __syncthreads();

  v8f acc2[8];
#pragma unroll
  for (int j = 0; j < 8; ++j) acc2[j] = splat8(lb[j * 16 + ln]);
  gemm_strip<128, 8>(tile, 128, lwt, 128, m0, lane, acc2);
#pragma unroll
  for (int j = 0; j < 8; ++j)
#pragma unroll
    for (int r = 0; r < 8; ++r) {
      size_t idx = (size_t)(t0 + m0 + hi8 + r) * 128 + j * 16 + ln;
      float v = h[idx] + acc2[j][r];
      h[idx] = v; hb[idx] = (__bf16)v;
    }
}

// ---------------------------------------------------------------------------
// Output head: ssp(h@out_w1 + b1) @ out_w2 + b2, segment-sum by graph.
// ---------------------------------------------------------------------------
__global__ void __launch_bounds__(256)
out_head(const __bf16* hb, const __bf16* ow1t, const float* ob1,
         const float* ow2, const float* ob2, const int* batch, float* gacc) {
  __shared__ float s[128 * 64];
  const int tid = threadIdx.x, lane = tid & 31, wave = tid >> 5;
  const int m0 = wave * 16, ln = lane & 15, hi8 = (lane >> 4) << 3;
  const int t0 = blockIdx.x * 128;

  v8f acc[4];
#pragma unroll
  for (int j = 0; j < 4; ++j) acc[j] = splat8(ob1[j * 16 + ln]);
  gemm_strip<128, 4>(hb + (size_t)t0 * 128, 128, ow1t, 128, m0, lane, acc);
#pragma unroll
  for (int j = 0; j < 4; ++j)
#pragma unroll
    for (int r = 0; r < 8; ++r)
      s[(m0 + hi8 + r) * 64 + j * 16 + ln] = ssp(acc[j][r]);
  __syncthreads();

  if (tid < 128) {
    int node = t0 + tid;
    float d = ob2[0];
#pragma unroll 8
    for (int j = 0; j < 64; ++j) d += s[tid * 64 + j] * ow2[j];
    atomicAdd(&gacc[batch[node]], d);
  }
}

__global__ void finalize(const float* gacc, float* out, int ng) {
  int i = threadIdx.x;
  if (i < ng) out[i] = 1.f / (1.f + __expf(-gacc[i]));
}

// ---------------------------------------------------------------------------
extern "C" void kernel_launch(void* const* d_in, const int* in_sizes, int n_in,
                              void* d_out, int out_size, void* d_ws, size_t ws_size,
                              hipStream_t stream) {
  const int N_ = in_sizes[0];       // 16384
  const int E_ = in_sizes[2] / 2;   // 262144
  const int L_ = 6;

  const int*   z    = (const int*)  d_in[0];
  const float* pos  = (const float*)d_in[1];
  const int*   ei   = (const int*)  d_in[2];
  const int*   bat  = (const int*)  d_in[3];
  const float* emb  = (const float*)d_in[4];
  const float* mw1  = (const float*)d_in[5];
  const float* mb1  = (const float*)d_in[6];
  const float* mw2  = (const float*)d_in[7];
  const float* mb2  = (const float*)d_in[8];
  const float* cw1  = (const float*)d_in[9];
  const float* cw2  = (const float*)d_in[10];
  const float* cb2  = (const float*)d_in[11];
  const float* lw   = (const float*)d_in[12];
  const float* lb   = (const float*)d_in[13];
  const float* ow1  = (const float*)d_in[14];
  const float* ob1  = (const float*)d_in[15];
  const float* ow2  = (const float*)d_in[16];
  const float* ob2  = (const float*)d_in[17];
  const int* rowi = ei;
  const int* coli = ei + E_;

  // ---- workspace carve (all sizes 256B-aligned) ----
  size_t off = 0;
  char* base = (char*)d_ws;
  auto carve = [&](size_t bytes) -> void* {
    void* p = base + off; off += (bytes + 255) & ~(size_t)255; return p;
  };
  __bf16* ea   = (__bf16*)carve((size_t)E_ * 64 * 2);
  float*  h    = (float*) carve((size_t)N_ * 128 * 4);
  __bf16* hb   = (__bf16*)carve((size_t)N_ * 128 * 2);
  float*  xh   = (float*) carve((size_t)N_ * 128 * 4);
  float*  agg  = (float*) carve((size_t)N_ * 128 * 4);
  float*  Cb   = (float*) carve((size_t)E_ * 4);
  __bf16* w1t  = (__bf16*)carve((size_t)L_ * 128 * 64 * 2);
  __bf16* w2t  = (__bf16*)carve((size_t)L_ * 128 * 128 * 2);
  __bf16* cf1t = (__bf16*)carve((size_t)L_ * 128 * 128 * 2);
  __bf16* cf2t = (__bf16*)carve((size_t)L_ * 128 * 128 * 2);
  __bf16* lwt  = (__bf16*)carve((size_t)L_ * 128 * 128 * 2);
  __bf16* ow1t = (__bf16*)carve((size_t)64 * 128 * 2);
  float*  gacc = (float*) carve(64);
  (void)ws_size; (void)n_in;

  // ---- one-time (per call) prep ----
  {
    int tot = 6 * 128 * 64 + 4 * (6 * 128 * 128) + 64 * 128;
    prep_weights<<<(tot + 255) / 256, 256, 0, stream>>>(
        mw1, mw2, cw1, cw2, lw, ow1, w1t, w2t, cf1t, cf2t, lwt, ow1t);
  }
  emb_init<<<(N_ * 128 + 255) / 256, 256, 0, stream>>>(z, emb, h, hb, N_ * 128);
  edge_geom<<<(E_ + 255) / 256, 256, 0, stream>>>(pos, rowi, coli, ea, Cb, E_);
  zero_f<<<1, 64, 0, stream>>>(gacc, 16);

  // ---- interaction layers ----
  for (int l = 0; l < L_; ++l) {
    node_lin1<<<N_ / 128, 256, 0, stream>>>(hb, cf1t + (size_t)l * 128 * 128, xh);
    zero_f<<<(N_ * 128 + 255) / 256, 256, 0, stream>>>(agg, N_ * 128);
    edge_filter_scatter<<<E_ / 128, 256, 0, stream>>>(
        ea, Cb,
        w1t + (size_t)l * 128 * 64,  mb1 + (size_t)l * 128,
        w2t + (size_t)l * 128 * 128, mb2 + (size_t)l * 128,
        xh, rowi, coli, agg);
    node_update<<<N_ / 128, 256, 0, stream>>>(
        agg, cf2t + (size_t)l * 128 * 128, cb2 + (size_t)l * 128,
        lwt + (size_t)l * 128 * 128, lb + (size_t)l * 128, h, hb);
  }

  // ---- readout ----
  out_head<<<N_ / 128, 256, 0, stream>>>(hb, ow1t, ob1, ow2, ob2, bat, gacc);
  finalize<<<1, 64, 0, stream>>>(gacc, (float*)d_out, out_size);
}
